// LeafClassifier_29764123361884
// MI455X (gfx1250) — matrix operand order repaired and checked by reading, not compiled, not run on hardware
//
#include <hip/hip_runtime.h>

#define L_SEQ 2048
#define BATCH 32
#define DIM_IN 1024
#define DENSE 256
#define HID 128
#define NTAG 4
#define IMPOSS -10000.0f

typedef __attribute__((ext_vector_type(16))) __bf16 v16bf;
typedef __attribute__((ext_vector_type(8)))  float v8f;

union Frag16 { v16bf v; unsigned u[8]; };
union Pre64 { uint4 q[4]; unsigned short h[32]; };  // one lane's 64B pre chunk

__device__ __forceinline__ unsigned short f32_to_bf16(float f) {
  unsigned u = __float_as_uint(f);
  return (unsigned short)((u + 0x7fffu + ((u >> 16) & 1u)) >> 16);
}
__device__ __forceinline__ float bf16_to_f32(unsigned short h) {
  return __uint_as_float(((unsigned)h) << 16);
}
__device__ __forceinline__ unsigned short to_bf16(float f) { return f32_to_bf16(f); }
__device__ __forceinline__ unsigned short to_bf16(unsigned short h) { return h; }
__device__ __forceinline__ v8f zero8() {
  v8f z = {0.f, 0.f, 0.f, 0.f, 0.f, 0.f, 0.f, 0.f};
  return z;
}

// fast device math: v_rcp_f32, and hw v_tanh_f32 when the builtin exists
__device__ __forceinline__ float fast_rcp(float x) { return __builtin_amdgcn_rcpf(x); }
__device__ __forceinline__ float fast_sigmoid(float x) {
  return fast_rcp(1.f + __expf(-x));
}
__device__ __forceinline__ float fast_tanh(float x) {
#if __has_builtin(__builtin_amdgcn_tanhf)
  return __builtin_amdgcn_tanhf(x);
#elif __has_builtin(__builtin_amdgcn_tanh_f32)
  return __builtin_amdgcn_tanh_f32(x);
#else
  return __builtin_fmaf(2.f, fast_rcp(1.f + __expf(-2.f * x)), -1.f);
#endif
}

// ---------------------------------------------------------------------------
// f32 -> bf16 conversion (for recurrent weights)
// ---------------------------------------------------------------------------
__global__ __launch_bounds__(256) void cvt_bf16_kernel(const float* __restrict__ in,
                                                       unsigned short* __restrict__ out,
                                                       int n) {
  int i = blockIdx.x * blockDim.x + threadIdx.x;
  if (i < n) out[i] = f32_to_bf16(in[i]);
}

// ---------------------------------------------------------------------------
// masks[b][t] = (sum_d src[t][b][d] > 0)  -- one wave per (t,b) row
// ---------------------------------------------------------------------------
__global__ __launch_bounds__(256) void mask_kernel(const float* __restrict__ src,
                                                   float* __restrict__ masks) {
  const int gid = blockIdx.x * blockDim.x + threadIdx.x;
  const int row = gid >> 5;            // (t * BATCH + b)
  const int lane = threadIdx.x & 31;
  if (row >= L_SEQ * BATCH) return;
  const float* p = src + (size_t)row * DIM_IN;
  float s = 0.f;
  for (int i = lane; i < DIM_IN; i += 32) s += p[i];
#pragma unroll
  for (int off = 16; off > 0; off >>= 1) s += __shfl_down(s, off, 32);
  if (lane == 0) {
    const int t = row / BATCH, b = row % BATCH;
    masks[(size_t)b * L_SEQ + t] = (s > 0.f) ? 1.f : 0.f;
  }
}

// ---------------------------------------------------------------------------
// C[M,N] = act(A[M,K] * W[N,K]^T + bias1 + bias2)   (bf16 WMMA, f32 accum)
// BM=128 BN=64 BK=32, 256 threads (8 waves, 4x2 wave grid, 2x2 tiles/wave)
// PERM=true writes the LSTM-friendly layout pre[t][j/16][j%16][b][gate]
// ---------------------------------------------------------------------------
template <typename InT, bool RELU, bool PERM>
__global__ __launch_bounds__(256) void gemm_bias_kernel(
    const InT* __restrict__ A, const float* __restrict__ W,
    const float* __restrict__ bias1, const float* __restrict__ bias2,
    unsigned short* __restrict__ Cout, int M, int N, int K) {
  constexpr int BM = 128, BN = 64, BK = 32, SA = BK + 8;  // pad: 80B row stride
  __shared__ alignas(16) unsigned short Alds[BM * SA];
  __shared__ alignas(16) unsigned short Blds[BN * SA];

  const int tid = threadIdx.x;
  const int wave = tid >> 5, lane = tid & 31;
  const int half = lane >> 4, lm = lane & 15;
  const int wm = (wave >> 1) * 32;     // 0,32,64,96
  const int wn = (wave & 1) * 32;      // 0,32
  const int bm = blockIdx.x * BM;
  const int bn = blockIdx.y * BN;

  v8f acc[2][2];
#pragma unroll
  for (int m = 0; m < 2; ++m)
#pragma unroll
    for (int n = 0; n < 2; ++n) acc[m][n] = zero8();

  for (int k0 = 0; k0 < K; k0 += BK) {
    {  // stage A tile (128x32), converting to bf16: 16 elems/thread
      const int r = tid >> 1, c = (tid & 1) * 16;
      const InT* ap = (const InT*)__builtin_assume_aligned(
          A + (size_t)(bm + r) * K + k0 + c, 16);
      unsigned short* dst = &Alds[r * SA + c];
#pragma unroll
      for (int i = 0; i < 16; ++i) dst[i] = to_bf16(ap[i]);
    }
    {  // stage W tile (64x32) from f32: 8 elems/thread
      const int r = tid >> 2, c = (tid & 3) * 8;
      const float* wpd = (const float*)__builtin_assume_aligned(
          W + (size_t)(bn + r) * K + k0 + c, 16);
      unsigned short* dst = &Blds[r * SA + c];
#pragma unroll
      for (int i = 0; i < 8; ++i) dst[i] = f32_to_bf16(wpd[i]);
    }
    __syncthreads();

    Frag16 af[2], bf[2];
#pragma unroll
    for (int mt = 0; mt < 2; ++mt) {
      const unsigned* p0 = (const unsigned*)&Alds[(wm + mt * 16 + lm) * SA + half * 16];
      const unsigned* p1 = (const unsigned*)&Alds[(wm + mt * 16 + lm) * SA + half * 16 + 8];
#pragma unroll
      for (int i = 0; i < 4; ++i) { af[mt].u[i] = p0[i]; af[mt].u[4 + i] = p1[i]; }
    }
#pragma unroll
    for (int nt = 0; nt < 2; ++nt) {
      const unsigned* p = (const unsigned*)&Blds[(wn + nt * 16 + lm) * SA + half * 16];
#pragma unroll
      for (int i = 0; i < 8; ++i) bf[nt].u[i] = p[i];
    }
#pragma unroll
    for (int mt = 0; mt < 2; ++mt)
#pragma unroll
      for (int nt = 0; nt < 2; ++nt)
        acc[mt][nt] = __builtin_amdgcn_wmma_f32_16x16x32_bf16(
            false, af[mt].v, false, bf[nt].v, (short)0, acc[mt][nt], false, false);
    __syncthreads();
  }

#pragma unroll
  for (int nt = 0; nt < 2; ++nt) {
    const int col = bn + wn + nt * 16 + lm;
    float bsum = bias1[col];
    if (bias2) bsum += bias2[col];
#pragma unroll
    for (int mt = 0; mt < 2; ++mt)
#pragma unroll
      for (int v = 0; v < 8; ++v) {
        const int row = bm + wm + mt * 16 + half * 8 + v;
        float val = acc[mt][nt][v] + bsum;
        if (RELU) val = fmaxf(val, 0.f);
        if (PERM) {
          // rows are (t*32+b); cols are gate*128+j. Write [t][j>>4][j&15][b][gate].
          const int t = row >> 5, b = row & 31;
          const int gate = col >> 7, jj = col & 127;
          const size_t idx =
              ((((size_t)t * 8 + (jj >> 4)) * 16 + (jj & 15)) * 32 + b) * 4 + gate;
          Cout[idx] = f32_to_bf16(val);
        } else {
          Cout[(size_t)row * N + col] = f32_to_bf16(val);
        }
      }
  }
}

// ---------------------------------------------------------------------------
// Bidirectional LSTM recurrence. grid.x = 2 (dir), 256 threads = 8 waves.
// Wave w owns j-chunk [w*16, w*16+16) across ALL four gates, both M-tiles:
// all gate values of a (b,j) pair land in the same lane.
// pre is in permuted layout [t][jc][lm][b][gate] -> per-lane 64B chunks.
// Double-buffered h in LDS -> single barrier per step; incremental pointers
// (no per-step address rebuild); hw tanh / v_rcp in the gate phase;
// next step's pre slice prefetched (global_prefetch_b8).
// ---------------------------------------------------------------------------
__global__ __launch_bounds__(256) void lstm_bidir_kernel(
    const unsigned short* __restrict__ pre_f, const unsigned short* __restrict__ pre_b,
    const unsigned short* __restrict__ whh_f, const unsigned short* __restrict__ whh_b,
    unsigned short* __restrict__ hf, unsigned short* __restrict__ hb) {
  const int dir = blockIdx.x;
  const unsigned short* __restrict__ pre = dir ? pre_b : pre_f;
  const unsigned short* __restrict__ whh = dir ? whh_b : whh_f;
  unsigned short* __restrict__ hout = dir ? hb : hf;

  __shared__ alignas(16) unsigned short hlds[2][BATCH * HID];

  const int tid = threadIdx.x;
  const int wave = tid >> 5, lane = tid & 31;
  const int half = lane >> 4, lm = lane & 15;
  const int j = wave * 16 + lm;  // hidden index owned by this lane

  for (int i = tid; i < BATCH * HID; i += 256) hlds[0][i] = 0;
  float cst[2][8];
#pragma unroll
  for (int m = 0; m < 2; ++m)
#pragma unroll
    for (int v = 0; v < 8; ++v) cst[m][v] = 0.f;

  // incremental pointers: pre row (256B per lane per step) and h output row
  const long PSTRIDE = 8 * 16 * 128;          // ushorts per timestep in pre
  const long HSTRIDE = (long)HID * BATCH;     // ushorts per timestep in h
  const long pd = dir ? -PSTRIDE : PSTRIDE;
  const long hd = dir ? -HSTRIDE : HSTRIDE;
  const int t0 = dir ? (L_SEQ - 1) : 0;
  const unsigned short* pp =
      pre + (size_t)t0 * PSTRIDE + ((size_t)wave * 16 + lm) * 128;
  unsigned short* hp = hout + (size_t)t0 * HSTRIDE + (size_t)j * BATCH;

  __syncthreads();

  for (int s = 0; s < L_SEQ; ++s) {
    const unsigned short* hbuf = hlds[s & 1];
    unsigned short* hnxt = hlds[(s & 1) ^ 1];

    // prefetch next timestep's pre slice (speculative; adjacent ws is valid)
    __builtin_prefetch((const char*)(pp + pd), 0, 1);
    __builtin_prefetch((const char*)(pp + pd) + 128, 0, 1);

    // issue this step's pre loads early: 2 x 64B contiguous per lane
    Pre64 pch[2];
#pragma unroll
    for (int m = 0; m < 2; ++m) {
      const uint4* q = (const uint4*)__builtin_assume_aligned(
          pp + (m * 16 + half * 8) * 4, 16);
#pragma unroll
      for (int i = 0; i < 4; ++i) pch[m].q[i] = q[i];
    }

    v8f acc[2][4];
#pragma unroll
    for (int m = 0; m < 2; ++m)
#pragma unroll
      for (int g = 0; g < 4; ++g) acc[m][g] = zero8();

#pragma unroll
    for (int k0 = 0; k0 < HID; k0 += 32) {
      Frag16 af[2];
#pragma unroll
      for (int m = 0; m < 2; ++m) {
        const unsigned* p0 = (const unsigned*)&hbuf[(m * 16 + lm) * HID + k0 + half * 16];
        const unsigned* p1 = (const unsigned*)&hbuf[(m * 16 + lm) * HID + k0 + half * 16 + 8];
#pragma unroll
        for (int i = 0; i < 4; ++i) { af[m].u[i] = p0[i]; af[m].u[4 + i] = p1[i]; }
      }
      Frag16 bf[4];
#pragma unroll
      for (int g = 0; g < 4; ++g) {  // W_hh rows: n = g*HID + wave*16 + lm
        const unsigned* p = (const unsigned*)__builtin_assume_aligned(
            &whh[((size_t)(g * HID + wave * 16 + lm)) * HID + k0 + half * 16], 16);
#pragma unroll
        for (int i = 0; i < 8; ++i) bf[g].u[i] = p[i];
      }
#pragma unroll
      for (int m = 0; m < 2; ++m)
#pragma unroll
        for (int g = 0; g < 4; ++g)
          acc[m][g] = __builtin_amdgcn_wmma_f32_16x16x32_bf16(
              false, af[m].v, false, bf[g].v, (short)0, acc[m][g], false, false);
    }

    unsigned short hnew[2][8];
#pragma unroll
    for (int m = 0; m < 2; ++m) {
#pragma unroll
      for (int v = 0; v < 8; ++v) {
        float ig = acc[m][0][v] + bf16_to_f32(pch[m].h[v * 4 + 0]);
        float fg = acc[m][1][v] + bf16_to_f32(pch[m].h[v * 4 + 1]);
        float gg = acc[m][2][v] + bf16_to_f32(pch[m].h[v * 4 + 2]);
        float og = acc[m][3][v] + bf16_to_f32(pch[m].h[v * 4 + 3]);
        float c = fast_sigmoid(fg) * cst[m][v] + fast_sigmoid(ig) * fast_tanh(gg);
        cst[m][v] = c;
        float h = fast_sigmoid(og) * fast_tanh(c);
        hnew[m][v] = f32_to_bf16(h);
      }
      // packed h store: h_perm[t][j][b], 8 consecutive b = one b128 store
      uint4 pk;
      pk.x = (unsigned)hnew[m][0] | ((unsigned)hnew[m][1] << 16);
      pk.y = (unsigned)hnew[m][2] | ((unsigned)hnew[m][3] << 16);
      pk.z = (unsigned)hnew[m][4] | ((unsigned)hnew[m][5] << 16);
      pk.w = (unsigned)hnew[m][6] | ((unsigned)hnew[m][7] << 16);
      *(uint4*)__builtin_assume_aligned(hp + m * 16 + half * 8, 16) = pk;
    }

    // write new h into the other LDS buffer (no WAR hazard -> one barrier)
#pragma unroll
    for (int m = 0; m < 2; ++m)
#pragma unroll
      for (int v = 0; v < 8; ++v) {
        const int b = m * 16 + half * 8 + v;
        hnxt[b * HID + j] = hnew[m][v];
      }
    __syncthreads();  // new h visible to all waves

    pp += pd;
    hp += hd;
  }
}

// ---------------------------------------------------------------------------
// emit[t][b][c] = feats . crf_w[c] + crf_b[c], feats = [hf | hb] (permuted
// layout [t][j][b]). One wave per timestep, lane = batch -> coalesced reads,
// crf_w reads are lane-uniform (scalarized by the compiler).
// ---------------------------------------------------------------------------
__global__ __launch_bounds__(256) void emit_kernel(
    const unsigned short* __restrict__ hfp, const unsigned short* __restrict__ hbp,
    const float* __restrict__ crf_w, const float* __restrict__ crf_b,
    float* __restrict__ emitv) {
  const int wv = (blockIdx.x * blockDim.x + threadIdx.x) >> 5;  // = t
  const int b = threadIdx.x & 31;
  if (wv >= L_SEQ) return;
  const unsigned short* pf = hfp + (size_t)wv * HID * BATCH + b;
  const unsigned short* pb = hbp + (size_t)wv * HID * BATCH + b;
  float s[NTAG];
#pragma unroll
  for (int c = 0; c < NTAG; ++c) s[c] = crf_b[c];
  for (int jj = 0; jj < HID; ++jj) {
    const float vf = bf16_to_f32(pf[(size_t)jj * BATCH]);
    const float vb = bf16_to_f32(pb[(size_t)jj * BATCH]);
#pragma unroll
    for (int c = 0; c < NTAG; ++c)
      s[c] += vf * crf_w[c * (2 * HID) + jj] + vb * crf_w[c * (2 * HID) + HID + jj];
  }
#pragma unroll
  for (int c = 0; c < NTAG; ++c)
    emitv[((size_t)wv * BATCH + b) * NTAG + c] = s[c];
}

// ---------------------------------------------------------------------------
// Viterbi decode: 1 block, 1 thread per batch item. Writes best_score + tags.
// ---------------------------------------------------------------------------
__global__ void viterbi_kernel(const float* __restrict__ emitv,
                               const float* __restrict__ masks,
                               const float* __restrict__ trans,
                               unsigned char* __restrict__ bp,
                               float* __restrict__ out) {
  const int b = threadIdx.x;
  if (b >= BATCH) return;
  float tr[NTAG][NTAG];
#pragma unroll
  for (int c = 0; c < NTAG; ++c)
#pragma unroll
    for (int p = 0; p < NTAG; ++p) tr[c][p] = trans[c * NTAG + p];

  float score[NTAG] = {IMPOSS, IMPOSS, 0.0f, IMPOSS};  // START = NTAG-2 = 2
  for (int t = 0; t < L_SEQ; ++t) {
    const float m = masks[(size_t)b * L_SEQ + t];
    const float* e = emitv + ((size_t)t * BATCH + b) * NTAG;
    float ns[NTAG];
#pragma unroll
    for (int c = 0; c < NTAG; ++c) {
      float best = score[0] + tr[c][0];
      int arg = 0;
#pragma unroll
      for (int p = 1; p < NTAG; ++p) {
        const float v = score[p] + tr[c][p];
        if (v > best) { best = v; arg = p; }
      }
      bp[((size_t)t * BATCH + b) * NTAG + c] = (unsigned char)arg;
      ns[c] = best + e[c];
    }
#pragma unroll
    for (int c = 0; c < NTAG; ++c) score[c] = ns[c] * m + score[c] * (1.f - m);
  }
  // final += transitions[STOP]; STOP = NTAG-1 = 3
  float best = score[0] + tr[NTAG - 1][0];
  int tag = 0;
#pragma unroll
  for (int c = 1; c < NTAG; ++c) {
    const float v = score[c] + tr[NTAG - 1][c];
    if (v > best) { best = v; tag = c; }
  }
  out[b] = best;
  float* tags = out + BATCH;  // (B, L) row-major
  tags[(size_t)b * L_SEQ + (L_SEQ - 1)] = (float)tag;
  for (int t = L_SEQ - 1; t >= 1; --t) {
    tag = bp[((size_t)t * BATCH + b) * NTAG + tag];
    tags[(size_t)b * L_SEQ + (t - 1)] = (float)tag;
  }
}

// ---------------------------------------------------------------------------
extern "C" void kernel_launch(void* const* d_in, const int* in_sizes, int n_in,
                              void* d_out, int out_size, void* d_ws, size_t ws_size,
                              hipStream_t stream) {
  (void)in_sizes; (void)n_in; (void)out_size; (void)ws_size;
  const float* src    = (const float*)d_in[0];
  const float* fc1_w  = (const float*)d_in[1];
  const float* fc1_b  = (const float*)d_in[2];
  const float* w_ih_f = (const float*)d_in[3];
  const float* w_hh_f = (const float*)d_in[4];
  const float* b_ih_f = (const float*)d_in[5];
  const float* b_hh_f = (const float*)d_in[6];
  const float* w_ih_b = (const float*)d_in[7];
  const float* w_hh_b = (const float*)d_in[8];
  const float* b_ih_b = (const float*)d_in[9];
  const float* b_hh_b = (const float*)d_in[10];
  const float* crf_w  = (const float*)d_in[11];
  const float* crf_b  = (const float*)d_in[12];
  const float* trans  = (const float*)d_in[13];

  char* wp = (char*)d_ws;
  auto take = [&](size_t bytes) {
    char* p = wp;
    wp += (bytes + 255) & ~(size_t)255;
    return p;
  };
  const size_t LB = (size_t)L_SEQ * BATCH;  // 65536 rows
  unsigned short* x      = (unsigned short*)take(LB * DENSE * 2);      // 32 MB bf16
  unsigned short* pre_f  = (unsigned short*)take(LB * 4 * HID * 2);    // 64 MB bf16
  unsigned short* pre_b  = (unsigned short*)take(LB * 4 * HID * 2);    // 64 MB bf16
  unsigned short* hf     = (unsigned short*)take(LB * HID * 2);        // 16 MB bf16
  unsigned short* hb     = (unsigned short*)take(LB * HID * 2);        // 16 MB bf16
  unsigned short* whhf16 = (unsigned short*)take((size_t)4 * HID * HID * 2);
  unsigned short* whhb16 = (unsigned short*)take((size_t)4 * HID * HID * 2);
  float* emitv           = (float*)take(LB * NTAG * 4);
  float* masks           = (float*)take(LB * 4);
  unsigned char* bp      = (unsigned char*)take(LB * NTAG);

  {  // recurrent weights -> bf16 once per call (deterministic)
    const int n = 4 * HID * HID;
    cvt_bf16_kernel<<<dim3((n + 255) / 256), dim3(256), 0, stream>>>(w_hh_f, whhf16, n);
    cvt_bf16_kernel<<<dim3((n + 255) / 256), dim3(256), 0, stream>>>(w_hh_b, whhb16, n);
  }
  mask_kernel<<<dim3((unsigned)(LB / 8)), dim3(256), 0, stream>>>(src, masks);

  // x = relu(src @ fc1_w^T + fc1_b)        M=65536 N=256 K=1024
  gemm_bias_kernel<float, true, false>
      <<<dim3((unsigned)(LB / 128), DENSE / 64), dim3(256), 0, stream>>>(
          src, fc1_w, fc1_b, nullptr, x, (int)LB, DENSE, DIM_IN);
  // pre = x @ w_ih^T + b_ih + b_hh (permuted layout)   M=65536 N=512 K=256
  gemm_bias_kernel<unsigned short, false, true>
      <<<dim3((unsigned)(LB / 128), (4 * HID) / 64), dim3(256), 0, stream>>>(
          x, w_ih_f, b_ih_f, b_hh_f, pre_f, (int)LB, 4 * HID, DENSE);
  gemm_bias_kernel<unsigned short, false, true>
      <<<dim3((unsigned)(LB / 128), (4 * HID) / 64), dim3(256), 0, stream>>>(
          x, w_ih_b, b_ih_b, b_hh_b, pre_b, (int)LB, 4 * HID, DENSE);

  lstm_bidir_kernel<<<dim3(2), dim3(256), 0, stream>>>(pre_f, pre_b, whhf16, whhb16, hf, hb);

  emit_kernel<<<dim3(L_SEQ / 8), dim3(256), 0, stream>>>(hf, hb, crf_w, crf_b, emitv);
  viterbi_kernel<<<dim3(1), dim3(32), 0, stream>>>(emitv, masks, trans, bp, (float*)d_out);
}
